// Network_91139206021653
// MI455X (gfx1250) — compile-verified, matrix operand order checked by reading, not run on hardware
//
#include <hip/hip_runtime.h>
#include <hip/hip_bf16.h>
#include <cstdint>

// ---------------------------------------------------------------------------
// MI455X (gfx1250) implementation of the FFT deconvolution network.
//  - Convolutions: V_WMMA_F32_16X16X32_F16 (M=16 outC, N=16 pixels, K tiles of 32),
//    input/weight tiles staged into LDS with GLOBAL_LOAD_ASYNC_TO_LDS_B32
//    (ASYNCcnt / s_wait_asynccnt), zero-halo via pre-zeroed LDS + EXEC-masked DMA.
//  - 2D FFT 320x320: in-place, mixed-radix 5*4*4*4 Stockham in LDS.
//    Rows pass: 2 transforms / block, coalesced.
//    Cols pass: tiled kernel, 8 adjacent columns / block, wave-aligned transforms,
//    row-major strip load (64B contiguous segments) -> no strided overfetch.
//  - Elementwise complex algebra as one-thread-per-element VALU kernels
// Workspace requirement: ~420 MB in d_ws (all big arrays are L2-resident: 52MB<192MB).
// ---------------------------------------------------------------------------

typedef __attribute__((ext_vector_type(16))) _Float16 v16h;
typedef __attribute__((ext_vector_type(8)))  float    v8f;

#define N_IMG 4
#define CCH   16
#define HV    256
#define WV    256
#define FFTN  320
#define PPLANE (FFTN*FFTN)
#define PAD4  30      // HK-1
#define CROP  15      // HK//2
#define KSCALE 100.0f
#define EPSV   1e-6f
#define CPB   8       // columns per block in the column-FFT pass

// ------------------------------- complex helpers ---------------------------
__device__ __forceinline__ float2 cmk(float a, float b){ return make_float2(a,b); }
__device__ __forceinline__ float2 cadd(float2 a,float2 b){ return cmk(a.x+b.x, a.y+b.y); }
__device__ __forceinline__ float2 csub(float2 a,float2 b){ return cmk(a.x-b.x, a.y-b.y); }
__device__ __forceinline__ float2 cmul(float2 a,float2 b){ return cmk(a.x*b.x-a.y*b.y, a.x*b.y+a.y*b.x); }
__device__ __forceinline__ float2 cmulc(float2 a,float2 b){ /* conj(a)*b */
  return cmk(a.x*b.x+a.y*b.y, a.x*b.y-a.y*b.x); }
__device__ __forceinline__ float2 cconj(float2 a){ return cmk(a.x,-a.y); }
__device__ __forceinline__ float2 cscl(float2 a,float s){ return cmk(a.x*s,a.y*s); }
__device__ __forceinline__ float  cabs2(float2 a){ return a.x*a.x+a.y*a.y; }
__device__ __forceinline__ float  softf(float x,float b){
  return copysignf(fmaxf(fabsf(x)-b,0.f), x); }

// ---- gfx1250 async LDS-DMA helpers (inline asm, per CDNA5 ISA) ---------
__device__ __forceinline__ void async_g2l_b32(uint32_t lds_addr, const void* gaddr){
#if defined(__gfx1250__)
  asm volatile("global_load_async_to_lds_b32 %0, %1, off"
               :: "v"(lds_addr), "v"(gaddr) : "memory");
#else
  (void)lds_addr; (void)gaddr;
#endif
}
__device__ __forceinline__ void wait_async0(){
#if defined(__gfx1250__)
  asm volatile("s_wait_asynccnt 0x0" ::: "memory");
#endif
}

// ------------------------------- weight prep -------------------------------
// wA0  : conv0 weights, centered + flipped, f16, [16][96]  (K = ci*25+r*5+s, zero padded)
// wAr  : rest weights, f16, [3][16][160]                    (K = ci*9 +r*3+s, zero padded)
// w0c  : swapaxes(centered weight0) f32 (3,16,5,5) for Fw0
__global__ void __launch_bounds__(256)
prep_weights(const float* __restrict__ w0, const float* __restrict__ wrest,
             _Float16* __restrict__ wA0, _Float16* __restrict__ wAr,
             float* __restrict__ w0c){
  int tid = threadIdx.x;
  for(int i=tid;i<16*96;i+=256)     wA0[i] = (_Float16)0.f;
  for(int i=tid;i<3*16*160;i+=256)  wAr[i] = (_Float16)0.f;
  __syncthreads();
  if (tid < 48){
    int co = tid/3, ci = tid%3;
    const float* f = w0 + ((size_t)co*3 + ci)*25;
    float m = 0.f;
    for(int i=0;i<25;++i) m += f[i];
    m *= (1.f/25.f);
    for(int r=0;r<5;++r) for(int s=0;s<5;++s){
      float wcv = f[r*5+s] - m;
      w0c[((size_t)ci*16 + co)*25 + r*5 + s] = wcv;            // swapaxes, unflipped
      wA0[co*96 + ci*25 + (4-r)*5 + (4-s)] = (_Float16)wcv;    // flipped for conv
    }
  }
  for(int i=tid;i<3*16*16*9;i+=256){
    int s = i%3; int rem = i/3; int r = rem%3; rem/=3;
    int ci = rem%16; rem/=16; int co = rem%16; int l = rem/16;
    wAr[((size_t)l*16 + co)*160 + ci*9 + r*3 + s] = (_Float16)wrest[i];
  }
}

// ------------------------------- WMMA conv ---------------------------------
// K-index mapping for f16 A/B fragments (per CDNA5 ISA 16-bit 16x32 layout):
// lane = half*16 + row ; VGPR j packs K pair:
__device__ __forceinline__ int kmapw(int j, int half){
  return (j < 4 ? 2*j : 16 + 2*(j-4)) + 8*half;
}

template<int CIN,int KH,int KW,int KPAD>
__global__ void __launch_bounds__(256)
conv_wmma(const float* __restrict__ in, const _Float16* __restrict__ wgt,
          float* __restrict__ out){
  const int TW = 128, IW = TW + KW - 1;
  __shared__ _Float16 sW[16*KPAD];
  __shared__ float    sIn[CIN*KH*IW];
  int tid = threadIdx.x;
  int n = blockIdx.z, y0 = blockIdx.y, x0 = blockIdx.x*TW;
  const int PH = KH/2, PW = KW/2;

#if defined(__gfx1250__)
  // ---- async LDS-DMA staging (ASYNCcnt path) ----
  {
    // weights: all dwords in-bounds -> pure DMA
    const uint32_t* wsrc = (const uint32_t*)wgt;
    uint32_t* wdst = (uint32_t*)&sW[0];
    for(int i=tid;i<(16*KPAD)/2;i+=256)
      async_g2l_b32((uint32_t)(uintptr_t)(wdst + i), wsrc + i);
  }
  // zero-fill input tile (halo default), then EXEC-masked async fill of interior
  for(int i=tid;i<CIN*KH*IW;i+=256) sIn[i] = 0.f;
  __syncthreads();                       // ds zeros complete before DMA writes land
  for(int i=tid;i<CIN*KH*IW;i+=256){
    int xx = i % IW; int rem = i / IW; int ky = rem % KH; int ci = rem / KH;
    int gy = y0 + ky - PH, gx = x0 + xx - PW;
    if (gy>=0 && gy<HV && gx>=0 && gx<WV){
      const float* g = &in[(((size_t)n*CIN+ci)*HV+gy)*WV+gx];
      async_g2l_b32((uint32_t)(uintptr_t)(&sIn[i]), g);
    }
  }
  // warm L2 for the next output row's tile while DMA drains
  if (y0 + 1 < HV)
    __builtin_prefetch(&in[(((size_t)n*CIN)*HV + (y0+1))*WV + x0], 0, 2);
  wait_async0();
  __syncthreads();
#else
  for(int i=tid;i<16*KPAD;i+=256) sW[i] = wgt[i];
  for(int i=tid;i<CIN*KH*IW;i+=256){
    int xx = i % IW; int rem = i / IW; int ky = rem % KH; int ci = rem / KH;
    int gy = y0 + ky - PH, gx = x0 + xx - PW;
    float v = 0.f;
    if (gy>=0 && gy<HV && gx>=0 && gx<WV)
      v = in[(((size_t)n*CIN+ci)*HV+gy)*WV+gx];
    sIn[i] = v;
  }
  __syncthreads();
#endif

  int lane = tid & 31, wave = tid >> 5;
  int half = lane >> 4, l16 = lane & 15;
  v8f acc = {};
  const int KT = KPAD/32;
  #pragma unroll
  for(int kt=0;kt<KT;++kt){
    v16h a, b;
    #pragma unroll
    for(int j=0;j<8;++j){
      int k0 = kmapw(j,half) + kt*32;
      a[2*j]   = sW[l16*KPAD + k0];
      a[2*j+1] = sW[l16*KPAD + k0 + 1];
      #pragma unroll
      for(int p=0;p<2;++p){
        int kk = k0 + p;
        _Float16 bv = (_Float16)0.f;
        if (kk < CIN*KH*KW){
          int ci = kk/(KH*KW); int rr = (kk/KW)%KH; int ss = kk%KW;
          bv = (_Float16)sIn[(ci*KH + rr)*IW + wave*16 + l16 + ss];
        }
        b[2*j+p] = bv;
      }
    }
    acc = __builtin_amdgcn_wmma_f32_16x16x32_f16(false, a, false, b,
                                                 (short)0, acc, false, false);
  }
  #pragma unroll
  for(int r=0;r<8;++r){
    int m = r + 8*half;
    out[(((size_t)n*16 + m)*HV + y0)*WV + (x0 + wave*16 + l16)] = acc[r];
  }
}

// ------------------------------- FFT 320 -----------------------------------
// Stockham autosort stage, generic radix, generic thread stride.
template<int R>
__device__ __forceinline__ void radix_stage(const float2* src, float2* dst,
                                            int t, int step, int Ns, float s){
  const int NR = FFTN / R;
  for(int j=t; j<NR; j+=step){
    float2 v[R];
    int jm = j % Ns;
    float ang0 = s * 6.2831853071795864f * (float)jm / (float)(Ns*R);
    #pragma unroll
    for(int r=0;r<R;++r){
      float2 x = src[j + r*NR];
      float sn, cs;
      __sincosf(ang0 * (float)r, &sn, &cs);
      v[r] = cmk(x.x*cs - x.y*sn, x.x*sn + x.y*cs);
    }
    int idxD = (j/Ns)*(Ns*R) + jm;
    #pragma unroll
    for(int k=0;k<R;++k){
      float2 acc = v[0];
      #pragma unroll
      for(int r=1;r<R;++r){
        float sn, cs;
        __sincosf(s * 6.2831853071795864f * (float)(r*k) / (float)R, &sn, &cs);
        acc.x += v[r].x*cs - v[r].y*sn;
        acc.y += v[r].x*sn + v[r].y*cs;
      }
      dst[idxD + Ns*k] = acc;
    }
  }
}

// Row pass: 2 transforms per 256-thread block (128 threads each); in-place.
__global__ void __launch_bounds__(256)
fft320(float2* data, int inverse){
  __shared__ float2 sA[2][FFTN];
  __shared__ float2 sB[2][FFTN];
  int tid = threadIdx.x;
  int sub = tid >> 7;
  int t   = tid & 127;
  int bi  = blockIdx.x*2 + sub;
  int plane = bi / FFTN;
  int r     = bi % FFTN;
  size_t base = (size_t)plane*PPLANE + (size_t)r*FFTN;
  for(int i=t;i<FFTN;i+=128) sA[sub][i] = data[base + i];
  __syncthreads();
  float s = inverse ? 1.f : -1.f;
  radix_stage<5>(sA[sub], sB[sub], t, 128, 1,  s); __syncthreads();
  radix_stage<4>(sB[sub], sA[sub], t, 128, 5,  s); __syncthreads();
  radix_stage<4>(sA[sub], sB[sub], t, 128, 20, s); __syncthreads();
  radix_stage<4>(sB[sub], sA[sub], t, 128, 80, s); __syncthreads();
  float sc = inverse ? (1.f/(float)FFTN) : 1.f;
  for(int i=t;i<FFTN;i+=128){
    float2 v = sA[sub][i];
    data[base + i] = cmk(v.x*sc, v.y*sc);
  }
}

// Column pass: 8 adjacent columns per block, wave-aligned transforms
// (column c = tid>>5, 32 threads per 320-point transform). Strip is loaded
// row-major (64B contiguous per row segment) -> no strided overfetch.
// LDS rows padded to 321 float2 so transform-phase accesses are conflict-free.
__global__ void __launch_bounds__(256)
fft320c(float2* data, int inverse){
  __shared__ float2 sA[CPB][FFTN+1];
  __shared__ float2 sB[CPB][FFTN+1];
  int tid = threadIdx.x;
  int plane = blockIdx.x / (FFTN/CPB);
  int x0 = (blockIdx.x % (FFTN/CPB)) * CPB;
  size_t base = (size_t)plane*PPLANE + x0;
  for(int idx=tid; idx<FFTN*CPB; idx+=256){
    int y = idx >> 3, c = idx & (CPB-1);
    sA[c][y] = data[base + (size_t)y*FFTN + c];
  }
  __syncthreads();
  float s = inverse ? 1.f : -1.f;
  int c  = tid >> 5;        // one wave per column transform
  int tt = tid & 31;
  radix_stage<5>(sA[c], sB[c], tt, 32, 1,  s); __syncthreads();
  radix_stage<4>(sB[c], sA[c], tt, 32, 5,  s); __syncthreads();
  radix_stage<4>(sA[c], sB[c], tt, 32, 20, s); __syncthreads();
  radix_stage<4>(sB[c], sA[c], tt, 32, 80, s); __syncthreads();
  float sc = inverse ? (1.f/(float)FFTN) : 1.f;
  for(int idx=tid; idx<FFTN*CPB; idx+=256){
    int y = idx >> 3, cc = idx & (CPB-1);
    float2 v = sA[cc][y];
    data[base + (size_t)y*FFTN + cc] = cmk(v.x*sc, v.y*sc);
  }
}

// ------------------------------- elementwise -------------------------------
__global__ void place_pad(const float* __restrict__ src, float* __restrict__ dst,
                          int planes){
  long i = (long)blockIdx.x*blockDim.x + threadIdx.x;
  long total = (long)planes*PPLANE;
  if (i >= total) return;
  int pix = (int)(i % PPLANE); long pl = i / PPLANE;
  int y = pix / FFTN, x = pix % FFTN;
  float v = 0.f;
  if (y>=PAD4 && y<PAD4+HV && x>=PAD4 && x<PAD4+WV)
    v = src[((size_t)pl*HV + (y-PAD4))*WV + (x-PAD4)];
  dst[i] = v;
}

__global__ void init_delta(float* k){
  int i = blockIdx.x*blockDim.x + threadIdx.x;
  if (i >= N_IMG*961) return;
  k[i] = ((i % 961) == 480) ? 1.f : 0.f;
}

__global__ void roll_soft(const float* __restrict__ wy3,
                          const float* __restrict__ biases,
                          float2* __restrict__ out){
  long i = (long)blockIdx.x*blockDim.x + threadIdx.x;
  long total = (long)N_IMG*CCH*PPLANE;
  if (i >= total) return;
  int pix = (int)(i % PPLANE); long pl = i / PPLANE; int c = (int)(pl % CCH);
  int y = pix / FFTN, x = pix % FFTN;
  int ys = (y + CROP) % FFTN, xs = (x + CROP) % FFTN;
  float v = wy3[(size_t)pl*PPLANE + (size_t)ys*FFTN + xs];
  out[i] = cmk(softf(v, biases[c]), 0.f);
}

__global__ void r2c(const float* __restrict__ src, float2* __restrict__ dst, long n){
  long i = (long)blockIdx.x*blockDim.x + threadIdx.x;
  if (i >= n) return;
  dst[i] = cmk(src[i], 0.f);
}

__global__ void embed_k(const float* __restrict__ k, float2* __restrict__ out){
  long i = (long)blockIdx.x*blockDim.x + threadIdx.x;
  if (i >= (long)N_IMG*PPLANE) return;
  int pix = (int)(i % PPLANE); int n = (int)(i / PPLANE);
  int y = pix / FFTN, x = pix % FFTN;
  float v = (y < 31 && x < 31) ? k[n*961 + y*31 + x] : 0.f;
  out[i] = cmk(v, 0.f);
}

__global__ void ew_fg(const float2* __restrict__ fk, const float2* __restrict__ ffy,
                      const float2* __restrict__ fz, const float* __restrict__ zetas,
                      int layer, float2* __restrict__ fg, float2* __restrict__ dup){
  long i = (long)blockIdx.x*blockDim.x + threadIdx.x;
  long total = (long)N_IMG*CCH*PPLANE;
  if (i >= total) return;
  int pix = (int)(i % PPLANE); long pl = i / PPLANE;
  int c = (int)(pl % CCH); int n = (int)(pl / CCH);
  float2 K = fk[(size_t)n*PPLANE + pix];
  float z = zetas[layer*CCH + c];                 // PROX_SCALE == 1
  float2 num = cadd(cscl(cmulc(K, ffy[i]), z), fz[i]);
  float inv = 1.f / (z * cabs2(K) + 1.f);
  float2 g = cscl(num, inv);
  fg[i] = g; dup[i] = g;
}

__global__ void soft_c(float2* __restrict__ buf, const float* __restrict__ biases,
                       int row){
  long i = (long)blockIdx.x*blockDim.x + threadIdx.x;
  long total = (long)N_IMG*CCH*PPLANE;
  if (i >= total) return;
  int c = (int)((i / PPLANE) % CCH);
  buf[i] = cmk(softf(buf[i].x, biases[row*CCH + c]), 0.f);
}

__global__ void knd_kernel(const float2* __restrict__ fz, const float2* __restrict__ ffy,
                           const float2* __restrict__ fk, const float* __restrict__ kprox,
                           int layer, float2* __restrict__ knd){
  long i = (long)blockIdx.x*blockDim.x + threadIdx.x;
  if (i >= (long)N_IMG*PPLANE) return;
  int pix = (int)(i % PPLANE); int n = (int)(i / PPLANE);
  float2 num = cmk(0.f,0.f); float den = 0.f;
  #pragma unroll
  for(int c=0;c<CCH;++c){
    size_t idx = ((size_t)n*CCH + c)*PPLANE + pix;
    float2 z = fz[idx], y = ffy[idx];
    num = cadd(num, cmulc(z, y));                 // conj(Fz)*Ffy
    den += cabs2(z);
  }
  float zk = kprox[layer];
  float2 K = fk[i];
  num = cadd(cscl(num, zk), K);
  float inv = 1.f / (zk*den + 1.f);
  knd[i] = cscl(num, inv);
}

__global__ void __launch_bounds__(256)
kpost(const float2* __restrict__ knd, const float* __restrict__ kbias,
      int layer, float* __restrict__ kout){
  __shared__ float red[256];
  __shared__ float kv[961];
  int n = blockIdx.x, tid = threadIdx.x;
  const float2* pl = knd + (size_t)n*PPLANE;
  float m = -3.4e38f;
  for(int i=tid;i<PPLANE;i+=256) m = fmaxf(m, pl[i].x);
  red[tid] = m; __syncthreads();
  for(int o=128;o>0;o>>=1){ if(tid<o) red[tid] = fmaxf(red[tid], red[tid+o]); __syncthreads(); }
  m = red[0]; __syncthreads();
  float se = 0.f;
  for(int i=tid;i<PPLANE;i+=256) se += __expf(KSCALE*(pl[i].x - m));
  red[tid] = se; __syncthreads();
  for(int o=128;o>0;o>>=1){ if(tid<o) red[tid] += red[tid+o]; __syncthreads(); }
  float kmax = m + __logf(red[0]) * (1.f/KSCALE);
  __syncthreads();
  float bk = kbias[layer];                        // KERNEL_BIAS_SCALE == 1
  for(int i=tid;i<961;i+=256){
    int y = i/31, x = i%31;
    kv[i] = fmaxf(pl[y*FFTN + x].x - bk*kmax, 0.f);
  }
  __syncthreads();
  float s = 0.f;
  for(int i=tid;i<961;i+=256) s += kv[i];
  red[tid] = s; __syncthreads();
  for(int o=128;o>0;o>>=1){ if(tid<o) red[tid] += red[tid+o]; __syncthreads(); }
  float inv = 1.f/(red[0] + EPSV);
  for(int i=tid;i<961;i+=256)
    kout[n*961+i] = (kv[i] + (i==480 ? EPSV : 0.f)) * inv;
}

__global__ void embed_blurred(const float* __restrict__ img, float2* __restrict__ out){
  long i = (long)blockIdx.x*blockDim.x + threadIdx.x;
  if (i >= (long)N_IMG*3*PPLANE) return;
  int pix = (int)(i % PPLANE); long pl = i / PPLANE;
  int y = pix / FFTN, x = pix % FFTN;
  float v = 0.f;
  if (y>=PAD4 && y<PAD4+HV && x>=PAD4 && x<PAD4+WV)
    v = img[((size_t)pl*HV + (y-PAD4))*WV + (x-PAD4)];
  out[i] = cmk(v, 0.f);
}

__global__ void embed_w0(const float* __restrict__ w0c, float2* __restrict__ out){
  long i = (long)blockIdx.x*blockDim.x + threadIdx.x;
  if (i >= (long)48*PPLANE) return;
  int pix = (int)(i % PPLANE); int pl = (int)(i / PPLANE);
  int y = pix / FFTN, x = pix % FFTN;
  int yy = (y + 2) % FFTN, xx = (x + 2) % FFTN;   // roll by (-2,-2)
  float v = (yy < 5 && xx < 5) ? w0c[(size_t)pl*25 + yy*5 + xx] : 0.f;
  out[i] = cmk(v, 0.f);
}

__global__ void image_coeffs(const float2* __restrict__ fyp, const float2* __restrict__ fgf,
                             const float2* __restrict__ fw0, const float2* __restrict__ fkp,
                             const float* __restrict__ eta, float2* __restrict__ fx){
  long i = (long)blockIdx.x*blockDim.x + threadIdx.x;
  if (i >= (long)N_IMG*PPLANE) return;
  int pix = (int)(i % PPLANE); int n = (int)(i / PPLANE);
  float2 K = fkp[i]; float K2 = cabs2(K); float2 Kc = cconj(K);
  float2 Yr = fyp[((size_t)n*3+0)*PPLANE + pix];
  float2 Yg = fyp[((size_t)n*3+1)*PPLANE + pix];
  float2 Yb = fyp[((size_t)n*3+2)*PPLANE + pix];
  float Crr = K2, Cgg = K2, Cbb = K2;
  float2 Crg = cmk(0,0), Crb = cmk(0,0), Cgb = cmk(0,0);
  float2 Br = cmul(Kc, Yr), Bg = cmul(Kc, Yg), Bb = cmul(Kc, Yb);
  for(int c=0;c<CCH;++c){
    float ec = eta[c];                            // PROX_SCALE == 1
    float2 wr = fw0[((size_t)0*CCH + c)*PPLANE + pix];
    float2 wg = fw0[((size_t)1*CCH + c)*PPLANE + pix];
    float2 wb = fw0[((size_t)2*CCH + c)*PPLANE + pix];
    float2 g  = fgf[((size_t)n*CCH + c)*PPLANE + pix];
    Crr += ec*cabs2(wr); Cgg += ec*cabs2(wg); Cbb += ec*cabs2(wb);
    Crg = cadd(Crg, cscl(cmulc(wr, wg), ec));
    Crb = cadd(Crb, cscl(cmulc(wr, wb), ec));
    Cgb = cadd(Cgb, cscl(cmulc(wg, wb), ec));
    Br  = cadd(Br,  cscl(cmulc(wr, g),  ec));
    Bg  = cadd(Bg,  cscl(cmulc(wg, g),  ec));
    Bb  = cadd(Bb,  cscl(cmulc(wb, g),  ec));
  }
  float aGb2 = cabs2(Cgb), aRb2 = cabs2(Crb), aRg2 = cabs2(Crg);
  float  Irr = Cgg*Cbb - aGb2;
  float  Igg = Crr*Cbb - aRb2;
  float  Ibb = Crr*Cgg - aRg2;
  float2 Irg = csub(cmulc(Cgb, Crb), cscl(Crg, Cbb));
  float2 Irb = csub(cmul(Crg, Cgb),  cscl(Crb, Cgg));
  float2 Igb = csub(cmulc(Crg, Crb), cscl(Cgb, Crr));
  float2 tcr = cmulc(cmul(Crg, Cgb), Crb);
  float den = Crr*Irr - Cgg*aRb2 - Cbb*aRg2 + 2.f*tcr.x;
  float inv = 1.f/den;
  float2 Fxr = cscl(cadd(cadd(cscl(Br, Irr), cmul(Irg, Bg)), cmul(Irb, Bb)), inv);
  float2 Fxg = cscl(cadd(cadd(cmul(cconj(Irg), Br), cscl(Bg, Igg)), cmul(Igb, Bb)), inv);
  float2 Fxb = cscl(cadd(cadd(cmul(cconj(Irb), Br), cmul(cconj(Igb), Bg)), cscl(Bb, Ibb)), inv);
  fx[((size_t)n*3+0)*PPLANE + pix] = Fxr;
  fx[((size_t)n*3+1)*PPLANE + pix] = Fxg;
  fx[((size_t)n*3+2)*PPLANE + pix] = Fxb;
}

__global__ void crop_out(const float2* __restrict__ fx, float* __restrict__ out){
  long i = (long)blockIdx.x*blockDim.x + threadIdx.x;
  if (i >= (long)N_IMG*3*HV*WV) return;
  int x = (int)(i % WV); int y = (int)((i / WV) % HV); long pl = i / ((long)HV*WV);
  out[i] = fx[(size_t)pl*PPLANE + (size_t)(y+CROP)*FFTN + (x+CROP)].x;
}

__global__ void copy_k(const float* __restrict__ k, float* __restrict__ out){
  int i = blockIdx.x*blockDim.x + threadIdx.x;
  if (i >= N_IMG*961) return;
  out[(long)N_IMG*3*HV*WV + i] = k[i];
}

// ------------------------------- host orchestration ------------------------
extern "C" void kernel_launch(void* const* d_in, const int* in_sizes, int n_in,
                              void* d_out, int out_size, void* d_ws, size_t ws_size,
                              hipStream_t stream){
  (void)in_sizes; (void)n_in; (void)out_size; (void)ws_size;
  const float* blurred = (const float*)d_in[0];
  const float* weight0 = (const float*)d_in[1];
  const float* wrest   = (const float*)d_in[2];
  const float* biases  = (const float*)d_in[3];
  const float* kbias   = (const float*)d_in[4];
  const float* kprox   = (const float*)d_in[5];
  const float* zetas   = (const float*)d_in[6];
  const float* eta     = (const float*)d_in[7];
  float* out = (float*)d_out;

  char* ws = (char*)d_ws;
  size_t off = 0;
  auto alloc = [&](size_t bytes)->void*{
    void* p = ws + off;
    off += (bytes + 255) & ~(size_t)255;
    return p;
  };
  _Float16* wA0 = (_Float16*)alloc(16*96*sizeof(_Float16));
  _Float16* wAr = (_Float16*)alloc(3*16*160*sizeof(_Float16));
  float*  w0c   = (float*)alloc(1200*sizeof(float));
  float*  convA = (float*)alloc((size_t)N_IMG*CCH*HV*WV*sizeof(float));
  float*  convB = (float*)alloc((size_t)N_IMG*CCH*HV*WV*sizeof(float));
  float*  WY[4];
  for(int i=0;i<4;++i) WY[i] = (float*)alloc((size_t)N_IMG*CCH*PPLANE*sizeof(float));
  float2* FFY = (float2*)alloc((size_t)N_IMG*CCH*PPLANE*sizeof(float2));
  float2* FZA = (float2*)alloc((size_t)N_IMG*CCH*PPLANE*sizeof(float2));
  float2* FZB = (float2*)alloc((size_t)N_IMG*CCH*PPLANE*sizeof(float2));
  float2* FG  = (float2*)alloc((size_t)N_IMG*CCH*PPLANE*sizeof(float2));
  float2* FKP = (float2*)alloc((size_t)N_IMG*PPLANE*sizeof(float2));
  float2* KND = (float2*)alloc((size_t)N_IMG*PPLANE*sizeof(float2));
  float*  KBUF= (float*)alloc((size_t)N_IMG*961*sizeof(float));
  float2* FYP = (float2*)alloc((size_t)N_IMG*3*PPLANE*sizeof(float2));
  float2* FX  = (float2*)alloc((size_t)N_IMG*3*PPLANE*sizeof(float2));
  float2* FW0 = (float2*)alloc((size_t)48*PPLANE*sizeof(float2));

  auto g1 = [](long n){ return dim3((unsigned)((n + 255)/256)); };
  auto fft2p = [&](float2* p, int planes, int inv){
    int b = planes * FFTN;                 // rows: always even here
    fft320 <<<dim3((unsigned)(b/2)),          dim3(256), 0, stream>>>(p, inv);
    fft320c<<<dim3((unsigned)(planes*(FFTN/CPB))), dim3(256), 0, stream>>>(p, inv);
  };
  const long NCP = (long)N_IMG*CCH*PPLANE;

  // ---- CNN front-end (WMMA convs) ----
  prep_weights<<<1,256,0,stream>>>(weight0, wrest, wA0, wAr, w0c);
  conv_wmma<3,5,5,96><<<dim3(2,HV,N_IMG),256,0,stream>>>(blurred, wA0, convA);
  place_pad<<<g1(NCP),256,0,stream>>>(convA, WY[0], N_IMG*CCH);
  conv_wmma<16,3,3,160><<<dim3(2,HV,N_IMG),256,0,stream>>>(convA, wAr + 0*16*160, convB);
  place_pad<<<g1(NCP),256,0,stream>>>(convB, WY[1], N_IMG*CCH);
  conv_wmma<16,3,3,160><<<dim3(2,HV,N_IMG),256,0,stream>>>(convB, wAr + 1*16*160, convA);
  place_pad<<<g1(NCP),256,0,stream>>>(convA, WY[2], N_IMG*CCH);
  conv_wmma<16,3,3,160><<<dim3(2,HV,N_IMG),256,0,stream>>>(convA, wAr + 2*16*160, convB);
  place_pad<<<g1(NCP),256,0,stream>>>(convB, WY[3], N_IMG*CCH);

  // ---- initial z / Fz, k = delta ----
  init_delta<<<g1(N_IMG*961),256,0,stream>>>(KBUF);
  roll_soft<<<g1(NCP),256,0,stream>>>(WY[3], biases, FZA);
  fft2p(FZA, N_IMG*CCH, 0);
  float2* Fz = FZA; float2* S = FZB;

  // ---- layer loop ----
  for(int l=0;l<4;++l){
    r2c<<<g1(NCP),256,0,stream>>>(WY[3-l], FFY, NCP);
    fft2p(FFY, N_IMG*CCH, 0);
    embed_k<<<g1((long)N_IMG*PPLANE),256,0,stream>>>(KBUF, FKP);
    fft2p(FKP, N_IMG, 0);
    ew_fg<<<g1(NCP),256,0,stream>>>(FKP, FFY, Fz, zetas, l, FG, S);
    fft2p(S, N_IMG*CCH, 1);                         // ifft2(Fg)
    soft_c<<<g1(NCP),256,0,stream>>>(S, biases, l+1);
    fft2p(S, N_IMG*CCH, 0);                         // Fz = fft2(soft(...))
    { float2* t = Fz; Fz = S; S = t; }
    knd_kernel<<<g1((long)N_IMG*PPLANE),256,0,stream>>>(Fz, FFY, FKP, kprox, l, KND);
    fft2p(KND, N_IMG, 1);                           // k = ifft2(num/den).real
    kpost<<<N_IMG,256,0,stream>>>(KND, kbias, l, KBUF);
  }

  // ---- final solve ----
  embed_blurred<<<g1((long)N_IMG*3*PPLANE),256,0,stream>>>(blurred, FYP);
  fft2p(FYP, N_IMG*3, 0);
  embed_k<<<g1((long)N_IMG*PPLANE),256,0,stream>>>(KBUF, FKP);
  fft2p(FKP, N_IMG, 0);
  embed_w0<<<g1((long)48*PPLANE),256,0,stream>>>(w0c, FW0);
  fft2p(FW0, 48, 0);
  hipMemcpyAsync(S, FG, (size_t)NCP*sizeof(float2), hipMemcpyDeviceToDevice, stream);
  fft2p(S, N_IMG*CCH, 1);
  soft_c<<<g1(NCP),256,0,stream>>>(S, biases, 5);
  fft2p(S, N_IMG*CCH, 0);                           // Fg final
  image_coeffs<<<g1((long)N_IMG*PPLANE),256,0,stream>>>(FYP, S, FW0, FKP, eta, FX);
  fft2p(FX, N_IMG*3, 1);
  crop_out<<<g1((long)N_IMG*3*HV*WV),256,0,stream>>>(FX, out);
  copy_k<<<g1(N_IMG*961),256,0,stream>>>(KBUF, out);
}
